// InfiniteContextModel_6992206758354
// MI455X (gfx1250) — matrix operand order repaired and checked by reading, not compiled, not run on hardware
//
#include <hip/hip_runtime.h>
#include <hip/hip_bf16.h>
#include <math.h>

typedef __attribute__((ext_vector_type(16))) __bf16 v16bf;
typedef __attribute__((ext_vector_type(8)))  float  v8f;

constexpr int NB = 2;          // batch
constexpr int SQ = 4096;       // sequence
constexpr int HDIM = 1024;     // hidden
constexpr int NH = 16;         // heads
constexpr int HD = 64;         // head dim
constexpr int LMK = 64;        // landmarks
constexpr int MSLOTS = 512;    // memory slots
constexpr int TOPK = 32;       // == wave32 width, one lane per selected slot
constexpr int BS = NB * SQ;

static __device__ __forceinline__ unsigned short f32_to_bf16(float f) {
  unsigned u = __float_as_uint(f);
  unsigned r = 0x7FFFu + ((u >> 16) & 1u);
  return (unsigned short)((u + r) >> 16);
}
static __device__ __forceinline__ float wave_max32(float v) {
  for (int o = 16; o > 0; o >>= 1) v = fmaxf(v, __shfl_xor(v, o, 32));
  return v;
}
static __device__ __forceinline__ float wave_sum32(float v) {
  for (int o = 16; o > 0; o >>= 1) v += __shfl_xor(v, o, 32);
  return v;
}

// ---------------- f32 -> bf16 (same layout) ----------------
__global__ __launch_bounds__(256) void cvt_bf16_kernel(const float* __restrict__ in,
                                                       unsigned short* __restrict__ out, int n) {
  for (int i = blockIdx.x * 256 + threadIdx.x; i < n; i += gridDim.x * 256)
    out[i] = f32_to_bf16(in[i]);
}

// ---------------- f32 [K,N] -> bf16 [N,K] (transpose) ----------------
__global__ __launch_bounds__(256) void cvt_bf16_T_kernel(const float* __restrict__ W,
                                                         unsigned short* __restrict__ WT,
                                                         int K, int N) {
  int n_tot = K * N;
  for (int i = blockIdx.x * 256 + threadIdx.x; i < n_tot; i += gridDim.x * 256) {
    int k = i / N, n = i % N;
    WT[(size_t)n * K + k] = f32_to_bf16(W[i]);
  }
}

// ---------------- per-token L2 norm (one wave per token) ----------------
__global__ __launch_bounds__(256) void importance_kernel(const float* __restrict__ hs,
                                                         float* __restrict__ imp) {
  int lane = threadIdx.x & 31, wid = threadIdx.x >> 5;
  int t = blockIdx.x * 8 + wid;
  if (t >= BS) return;
  const float* row = hs + (size_t)t * HDIM;
  float s = 0.f;
  #pragma unroll
  for (int j = 0; j < HDIM / 32; ++j) { float x = row[j * 32 + lane]; s += x * x; }
  s = wave_sum32(s);
  if (lane == 0) imp[t] = sqrtf(s);
}

// ---------------- top-64 landmarks per batch, sorted ascending ----------------
__global__ __launch_bounds__(256) void topk_landmarks_kernel(const float* __restrict__ imp,
                                                             int* __restrict__ lmidx) {
  __shared__ float s_imp[SQ];
  __shared__ int   s_sel[LMK];
  __shared__ float s_wv[8];
  __shared__ int   s_wi[8];
  int b = blockIdx.x, tid = threadIdx.x;
  int lane = tid & 31, wid = tid >> 5;
  for (int i = tid; i < SQ; i += 256) s_imp[i] = imp[b * SQ + i];
  __syncthreads();
  for (int iter = 0; iter < LMK; ++iter) {
    float bv = -3.4e38f; int bi = 1 << 30;
    for (int i = tid; i < SQ; i += 256) {
      float v = s_imp[i];
      if (v > bv || (v == bv && i < bi)) { bv = v; bi = i; }
    }
    for (int o = 16; o > 0; o >>= 1) {
      float ov = __shfl_xor(bv, o, 32); int oi = __shfl_xor(bi, o, 32);
      if (ov > bv || (ov == bv && oi < bi)) { bv = ov; bi = oi; }
    }
    if (lane == 0) { s_wv[wid] = bv; s_wi[wid] = bi; }
    __syncthreads();
    if (tid == 0) {
      float fv = s_wv[0]; int fi = s_wi[0];
      for (int w = 1; w < 8; ++w)
        if (s_wv[w] > fv || (s_wv[w] == fv && s_wi[w] < fi)) { fv = s_wv[w]; fi = s_wi[w]; }
      s_sel[iter] = fi;
      s_imp[fi] = -3.4e38f;
    }
    __syncthreads();
  }
  if (tid == 0) {  // insertion sort of 64 indices
    for (int i = 1; i < LMK; ++i) {
      int key = s_sel[i], j = i - 1;
      while (j >= 0 && s_sel[j] > key) { s_sel[j + 1] = s_sel[j]; --j; }
      s_sel[j + 1] = key;
    }
  }
  __syncthreads();
  if (tid < LMK) lmidx[b * LMK + tid] = s_sel[tid];
}

// ---------------- K/V projection of the 64 landmark rows (f32, scalar) ----------------
// Output layout [b, head, l, d]
__global__ __launch_bounds__(256) void kv_proj_kernel(const float* __restrict__ hs,
                                                      const int* __restrict__ lmidx,
                                                      const float* __restrict__ Wk,
                                                      const float* __restrict__ bk,
                                                      const float* __restrict__ Wv,
                                                      const float* __restrict__ bvv,
                                                      float* __restrict__ kout,
                                                      float* __restrict__ vout) {
  __shared__ float s_h[HDIM];
  int blk = blockIdx.x, tid = threadIdx.x;
  int b = blk / LMK, l = blk % LMK;
  int tok = lmidx[b * LMK + l];
  const float* row = hs + ((size_t)(b * SQ + tok)) * HDIM;
  for (int i = tid; i < HDIM; i += 256) s_h[i] = row[i];
  __syncthreads();
  for (int j = tid; j < HDIM; j += 256) {
    float ak = bk[j], av = bvv[j];
    for (int h = 0; h < HDIM; ++h) {
      float x = s_h[h];
      ak += x * Wk[(size_t)h * HDIM + j];
      av += x * Wv[(size_t)h * HDIM + j];
    }
    int head = j >> 6, d = j & 63;
    size_t o = (((size_t)(b * NH + head)) * LMK + l) * HD + d;
    kout[o] = ak;
    vout[o] = av;
  }
}

// ---------------- WMMA bf16 GEMM: C = scale*(A @ B^T_rowmajor) + bias + addend ----------
// A: [M,K] bf16 row-major.  BT: [N,K] bf16 row-major (i.e. B pre-transposed).
// Block: 128 threads = 4 waves (2x2), block tile 64x128, wave tile 32x64.
// Per K-step: 8 WMMAs vs 12 x b128 loads (2 A-frags, 4 B-frags).
__global__ __launch_bounds__(128) void wmma_gemm_bf16(const unsigned short* __restrict__ Abf,
                                                      const unsigned short* __restrict__ BTbf,
                                                      const float* __restrict__ bias,
                                                      const float* __restrict__ addend,
                                                      float* __restrict__ Cout,
                                                      int Mdim, int Ndim, int Kdim,
                                                      float scale) {
  int lane = threadIdx.x & 31;
  int wid  = threadIdx.x >> 5;
  int rowBase = blockIdx.y * 64 + (wid >> 1) * 32;
  int colBase = blockIdx.x * 128 + (wid & 1) * 64;
  int l16 = lane & 15;
  int hA = (lane >> 4) * 8;    // A: K sub-offset per ISA 16-bit A layout
  int hB = (lane >> 4) * 8;    // B: K sub-offset per ISA 16-bit B layout

  const v8f vz = {0.f,0.f,0.f,0.f,0.f,0.f,0.f,0.f};
  v8f acc[2][4];
  #pragma unroll
  for (int mi = 0; mi < 2; ++mi)
    #pragma unroll
    for (int ni = 0; ni < 4; ++ni) acc[mi][ni] = vz;

  const unsigned short* ap0 = Abf + (size_t)(rowBase      + l16) * Kdim + hA;
  const unsigned short* ap1 = Abf + (size_t)(rowBase + 16 + l16) * Kdim + hA;
  const unsigned short* bp[4];
  #pragma unroll
  for (int ni = 0; ni < 4; ++ni)
    bp[ni] = BTbf + (size_t)(colBase + ni * 16 + l16) * Kdim + hB;

  for (int kb = 0; kb < Kdim; kb += 32) {
    union { v16bf v; uint4 q[2]; } a0, a1, bfr[4];
    a0.q[0] = *(const uint4*)(ap0 + kb);  a0.q[1] = *(const uint4*)(ap0 + kb + 16);
    a1.q[0] = *(const uint4*)(ap1 + kb);  a1.q[1] = *(const uint4*)(ap1 + kb + 16);
    #pragma unroll
    for (int ni = 0; ni < 4; ++ni) {
      bfr[ni].q[0] = *(const uint4*)(bp[ni] + kb);
      bfr[ni].q[1] = *(const uint4*)(bp[ni] + kb + 16);
    }
    if (kb + 32 < Kdim) {
      __builtin_prefetch(ap0 + kb + 32, 0, 1);
      __builtin_prefetch(ap1 + kb + 32, 0, 1);
    }
    #pragma unroll
    for (int ni = 0; ni < 4; ++ni) {
      acc[0][ni] = __builtin_amdgcn_wmma_f32_16x16x32_bf16(false, a0.v, false, bfr[ni].v,
                                                           (short)0, acc[0][ni], false, false);
      acc[1][ni] = __builtin_amdgcn_wmma_f32_16x16x32_bf16(false, a1.v, false, bfr[ni].v,
                                                           (short)0, acc[1][ni], false, false);
    }
  }

  // Epilogue per ISA 32-bit C/D layout: VGPR r -> row r (lanes 0-15) / row r+8 (lanes 16-31)
  #pragma unroll
  for (int mi = 0; mi < 2; ++mi) {
    #pragma unroll
    for (int ni = 0; ni < 4; ++ni) {
      int col = colBase + ni * 16 + l16;
      float badd = bias ? bias[col] : 0.f;
      #pragma unroll
      for (int r = 0; r < 8; ++r) {
        int row = rowBase + mi * 16 + (lane >> 4) * 8 + r;
        size_t idx = (size_t)row * Ndim + col;
        float v = scale * acc[mi][ni][r] + badd;
        if (addend) v += addend[idx];
        Cout[idx] = v;
      }
    }
  }
}

// ---------------- landmark attention: softmax(q.k/8) @ v per head ----------------
__global__ __launch_bounds__(256) void attention_kernel(const float* __restrict__ q,
                                                        const float* __restrict__ kmat,
                                                        const float* __restrict__ vmat,
                                                        unsigned short* __restrict__ ctx_bf) {
  __shared__ float s_k[LMK][HD + 1];   // padded: column-strided reads
  __shared__ float s_v[LMK][HD];       // unpadded: row-broadcast reads; async-loaded
  __shared__ float s_q[8][HD];
  __shared__ float s_attn[8][LMK];
  int bh = blockIdx.x >> 4;        // 16 chunks of 256 tokens each
  int chunk = blockIdx.x & 15;
  int b = bh >> 4, head = bh & 15;
  int tid = threadIdx.x, lane = tid & 31, wid = tid >> 5;

  size_t kvbase = ((size_t)(b * NH + head)) * LMK * HD;

  // V tile: CDNA5 async global->LDS copy, 16B per lane per iteration.
  {
    unsigned v_lds_base = (unsigned)(uintptr_t)(&s_v[0][0]);
    #pragma unroll
    for (int it = 0; it < (LMK * HD * 4) / (16 * 256); ++it) {
      int c = it * 256 + tid;                       // 16-byte chunk id
      const float* gsrc = vmat + kvbase + c * 4;
      unsigned ldst = v_lds_base + c * 16;
      asm volatile("global_load_async_to_lds_b128 %0, %1, off"
                   :: "v"(ldst), "v"(gsrc) : "memory");
    }
  }
  // K tile: regular loads into padded LDS.
  for (int i = tid; i < LMK * HD; i += 256)
    s_k[i >> 6][i & 63] = kmat[kvbase + i];
  asm volatile("s_wait_asynccnt 0x0" ::: "memory");
  __syncthreads();

  for (int t = 0; t < 32; ++t) {
    int s = chunk * 256 + t * 8 + wid;
    size_t qoff = ((size_t)(b * SQ + s)) * HDIM + head * HD;
    s_q[wid][lane]      = q[qoff + lane];
    s_q[wid][lane + 32] = q[qoff + lane + 32];

    int l0 = lane, l1 = lane + 32;
    float sc0 = 0.f, sc1 = 0.f;
    #pragma unroll
    for (int d = 0; d < HD; ++d) {
      float qd = s_q[wid][d];
      sc0 += qd * s_k[l0][d];
      sc1 += qd * s_k[l1][d];
    }
    sc0 *= 0.125f; sc1 *= 0.125f;
    float m = wave_max32(fmaxf(sc0, sc1));
    float e0 = __expf(sc0 - m), e1 = __expf(sc1 - m);
    float inv = 1.0f / wave_sum32(e0 + e1);
    s_attn[wid][l0] = e0 * inv;
    s_attn[wid][l1] = e1 * inv;

    float c0 = 0.f, c1 = 0.f;
    int d0 = lane, d1 = lane + 32;
    #pragma unroll
    for (int l = 0; l < LMK; ++l) {
      float a = s_attn[wid][l];
      c0 += a * s_v[l][d0];
      c1 += a * s_v[l][d1];
    }
    ctx_bf[qoff + d0] = f32_to_bf16(c0);
    ctx_bf[qoff + d1] = f32_to_bf16(c1);
  }
}

// ---------------- per-token top-32 over 512 memory slots + gated read ----------------
__global__ __launch_bounds__(256) void mem_topk_kernel(const float* __restrict__ mscores,
                                                       const float* __restrict__ hs,
                                                       const float* __restrict__ memv,
                                                       const float* __restrict__ gate_w,
                                                       const float* __restrict__ gate_b,
                                                       float* __restrict__ memout) {
  __shared__ float s_a[8][32];
  __shared__ int   s_g[8][32];
  int lane = threadIdx.x & 31, wid = threadIdx.x >> 5;
  int t = blockIdx.x * 8 + wid;
  if (t >= BS) return;

  float sc[16];
  const float* srow = mscores + (size_t)t * MSLOTS;
  #pragma unroll
  for (int j = 0; j < 16; ++j) sc[j] = srow[j * 32 + lane];

  float myv = 0.f; int myg = 0;
  for (int iter = 0; iter < TOPK; ++iter) {
    float bv = -3.4e38f; int bg = 1 << 30;
    #pragma unroll
    for (int j = 0; j < 16; ++j) {
      float v = sc[j];
      if (v > bv) { bv = v; bg = j * 32 + lane; }
    }
    for (int o = 16; o > 0; o >>= 1) {
      float ov = __shfl_xor(bv, o, 32); int og = __shfl_xor(bg, o, 32);
      if (ov > bv || (ov == bv && og < bg)) { bv = ov; bg = og; }
    }
    if (lane == iter) { myv = bv; myg = bg; }
    #pragma unroll
    for (int j = 0; j < 16; ++j)
      if (((bg & 31) == lane) && ((bg >> 5) == j)) sc[j] = -3.4e38f;
  }
  // softmax across the 32 lanes (lane i owns the i-th largest)
  float m = wave_max32(myv);
  float e = __expf(myv - m);
  float a = e / wave_sum32(e);
  s_a[wid][lane] = a;
  s_g[wid][lane] = myg;

  // gate = sigmoid(hs . gate_w + gate_b)
  const float* hrow = hs + (size_t)t * HDIM;
  float gsum = 0.f;
  #pragma unroll
  for (int j = 0; j < HDIM / 32; ++j) { int h = j * 32 + lane; gsum += hrow[h] * gate_w[h]; }
  gsum = wave_sum32(gsum) + gate_b[0];
  float gate = 1.f / (1.f + __expf(-gsum));

  float acc[32];
  #pragma unroll
  for (int j = 0; j < 32; ++j) acc[j] = 0.f;
  for (int i = 0; i < TOPK; ++i) {
    float ai = s_a[wid][i];
    const float* row = memv + (size_t)s_g[wid][i] * HDIM;
    #pragma unroll
    for (int j = 0; j < 32; ++j) acc[j] += ai * row[j * 32 + lane];
  }
  float* orow = memout + (size_t)t * HDIM;
  #pragma unroll
  for (int j = 0; j < 32; ++j) orow[j * 32 + lane] = gate * acc[j];
}

extern "C" void kernel_launch(void* const* d_in, const int* in_sizes, int n_in,
                              void* d_out, int out_size, void* d_ws, size_t ws_size,
                              hipStream_t stream) {
  (void)in_sizes; (void)n_in; (void)out_size; (void)ws_size;
  const float* hs   = (const float*)d_in[0];
  const float* Wq   = (const float*)d_in[1];
  const float* bq   = (const float*)d_in[2];
  const float* Wk   = (const float*)d_in[3];
  const float* bk   = (const float*)d_in[4];
  const float* Wv   = (const float*)d_in[5];
  const float* bvv  = (const float*)d_in[6];
  const float* Wo   = (const float*)d_in[7];
  const float* bo   = (const float*)d_in[8];
  const float* memk = (const float*)d_in[9];
  const float* memv = (const float*)d_in[10];
  const float* gw   = (const float*)d_in[11];
  const float* gb   = (const float*)d_in[12];
  // d_in[13] = top_k (always 32 for this problem; hardcoded as TOPK)
  float* out = (float*)d_out;

  char* base = (char*)d_ws;
  size_t off = 0;
  auto carve = [&](size_t bytes) -> void* {
    void* p = base + off;
    off = (off + bytes + 255) & ~(size_t)255;
    return p;
  };
  float*          imp     = (float*)carve((size_t)BS * 4);
  int*            lmidx   = (int*)carve((size_t)NB * LMK * 4);
  unsigned short* hs_bf   = (unsigned short*)carve((size_t)BS * HDIM * 2);
  unsigned short* WqT     = (unsigned short*)carve((size_t)HDIM * HDIM * 2);
  unsigned short* WoT     = (unsigned short*)carve((size_t)HDIM * HDIM * 2);
  unsigned short* memk_bf = (unsigned short*)carve((size_t)MSLOTS * HDIM * 2);
  float*          k_ws    = (float*)carve((size_t)NB * NH * LMK * HD * 4);
  float*          v_ws    = (float*)carve((size_t)NB * NH * LMK * HD * 4);
  float*          q_ws    = (float*)carve((size_t)BS * HDIM * 4);
  unsigned short* ctx_bf  = (unsigned short*)carve((size_t)BS * HDIM * 2);
  float*          mscores = (float*)carve((size_t)BS * MSLOTS * 4);
  float*          memout  = (float*)carve((size_t)BS * HDIM * 4);

  // 1) precision staging
  cvt_bf16_kernel<<<8192, 256, 0, stream>>>(hs, hs_bf, BS * HDIM);
  cvt_bf16_T_kernel<<<4096, 256, 0, stream>>>(Wq, WqT, HDIM, HDIM);
  cvt_bf16_T_kernel<<<4096, 256, 0, stream>>>(Wo, WoT, HDIM, HDIM);
  cvt_bf16_kernel<<<2048, 256, 0, stream>>>(memk, memk_bf, MSLOTS * HDIM);

  // 2) landmark selection
  importance_kernel<<<BS / 8, 256, 0, stream>>>(hs, imp);
  topk_landmarks_kernel<<<NB, 256, 0, stream>>>(imp, lmidx);

  // 3) K/V projection of landmarks (f32)
  kv_proj_kernel<<<NB * LMK, 256, 0, stream>>>(hs, lmidx, Wk, bk, Wv, bvv, k_ws, v_ws);

  // 4) Q projection (WMMA): q = hs @ Wq + bq
  wmma_gemm_bf16<<<dim3(HDIM / 128, BS / 64), 128, 0, stream>>>(
      hs_bf, WqT, bq, nullptr, q_ws, BS, HDIM, HDIM, 1.0f);

  // 5) landmark attention -> ctx (bf16 for the O-projection)
  attention_kernel<<<NB * NH * (SQ / 256), 256, 0, stream>>>(q_ws, k_ws, v_ws, ctx_bf);

  // 6) memory scores (WMMA): mscores = hs @ mem_keys^T / sqrt(H)
  wmma_gemm_bf16<<<dim3(MSLOTS / 128, BS / 64), 128, 0, stream>>>(
      hs_bf, memk_bf, nullptr, nullptr, mscores, BS, MSLOTS, HDIM, 0.03125f);

  // 7) top-32 + softmax + gated memory read
  mem_topk_kernel<<<BS / 8, 256, 0, stream>>>(mscores, hs, memv, gw, gb, memout);

  // 8) O projection (WMMA) fused with + bo + mem_out -> final output
  wmma_gemm_bf16<<<dim3(HDIM / 128, BS / 64), 128, 0, stream>>>(
      ctx_bf, WoT, bo, memout, out, BS, HDIM, HDIM, 1.0f);
}
